// SpatialCNN_small_23854248362245
// MI455X (gfx1250) — matrix operand order repaired and checked by reading, not compiled, not run on hardware
//
#include <hip/hip_runtime.h>
#include <hip/hip_bf16.h>
#include <math.h>

typedef __attribute__((ext_vector_type(16))) _Float16     v16h;
typedef __attribute__((ext_vector_type(8)))  float        v8f;
typedef __attribute__((ext_vector_type(8)))  unsigned int v8u;

#define EPSBN 1e-5f

// ---------------------------------------------------------------------------
// Pack conv weights (OIHW fp32, 3x3) -> fp16 [Co][Kpad], K = ci*9 + kh*3 + kw,
// zero padded to Kpad (multiple of 32) so the WMMA K-loop has no tail.
// ---------------------------------------------------------------------------
__global__ void k_pack_w(const float* __restrict__ w, _Float16* __restrict__ wp,
                         int Co, int Ci, int Kpad) {
  int i = blockIdx.x * blockDim.x + threadIdx.x;
  if (i >= Co * Kpad) return;
  int co = i / Kpad;
  int k  = i - co * Kpad;
  int Kt = Ci * 9;
  _Float16 val = (_Float16)0.f;
  if (k < Kt) val = (_Float16)w[(size_t)co * Kt + k];
  wp[i] = val;
}

// ---------------------------------------------------------------------------
// Implicit-GEMM conv3x3 (+dilation) with fused BN+ReLU using
// v_wmma_f32_16x16x32_f16.  Grid: x = wblk + nWb*coB, y = h, z = n.
// Block covers min(Co,64) output channels x 32 pixels along W at one (n,h).
// Waves: (coTile) x (pixTile 0..1).
// A = weights [16 co x 32 k] loaded straight from packed global weights,
// B = im2col patch [32 k x 16 pix] gathered from LDS, C/D = f32 accum.
// ---------------------------------------------------------------------------
__global__ void k_conv_wmma(const float* __restrict__ x,
                            const _Float16* __restrict__ wp,
                            const float* __restrict__ gw, const float* __restrict__ bw,
                            const float* __restrict__ mw, const float* __restrict__ vw,
                            float* __restrict__ out,
                            int Ci, int H, int W, int Co, int dil, int Kpad, int nWb) {
  __shared__ _Float16 Blds[32 * 576];  // 36.9 KB, max Kpad = 576 (Ci=64)

  int bx   = blockIdx.x;
  int wblk = bx % nWb;
  int coB  = bx / nWb;
  int h    = blockIdx.y;
  int n    = blockIdx.z;
  int x0   = wblk * 32;
  int co0  = coB * 64;
  int tid  = threadIdx.x;
  int Kt   = Ci * 9;

  // im2col stage: Blds[pix][k].  idx = k*32 + pix so the per-element index
  // math is shifts/masks plus one constant-divisor (/9) mul-hi.
  const float* xin = x + (size_t)n * Ci * H * W;
  for (int idx = tid; idx < Kt * 32; idx += blockDim.x) {
    int k   = idx >> 5;
    int pix = idx & 31;
    int ci  = k / 9;            // constant divisor -> mul-hi
    int r   = k - ci * 9;
    int kh  = r / 3;
    int kw  = r - kh * 3;
    int yy  = h + (kh - 1) * dil;
    int xx  = x0 + pix + (kw - 1) * dil;
    _Float16 val = (_Float16)0.f;
    if (yy >= 0 && yy < H && xx >= 0 && xx < W)
      val = (_Float16)xin[((size_t)ci * H + yy) * W + xx];
    Blds[pix * Kpad + k] = val;
  }
  // zero-fill the K padding rows
  for (int idx = tid; idx < (Kpad - Kt) * 32; idx += blockDim.x) {
    int k   = Kt + (idx >> 5);
    int pix = idx & 31;
    Blds[pix * Kpad + k] = (_Float16)0.f;
  }
  __syncthreads();

  int wv      = tid >> 5;
  int lane    = tid & 31;
  int coTile  = wv >> 1;
  int pixTile = wv & 1;
  int grp     = lane >> 4;   // lane half (selects K sub-range per ISA layout)
  int m       = lane & 15;   // row (A) / column (B) within the 16x16 tile

  int co = co0 + coTile * 16 + m;
  const _Float16* wRow = wp + (size_t)co * Kpad;
  const _Float16* bRow = &Blds[(size_t)(pixTile * 16 + m) * Kpad];

  v8f acc = {};
  for (int kc = 0; kc < Kpad; kc += 32) {
    // prefetch next weight chunk (global_prefetch_b8)
    __builtin_prefetch(wRow + kc + 32, 0, 1);
    union { v8u u; v16h h; } A, B;
#pragma unroll
    for (int t = 0; t < 8; ++t) {
      // A fragment: lanes 0-15 hold K {0..7,16..23}, lanes 16-31 {8..15,24..31}
      int ka = ((t < 4) ? (2 * t) : (2 * t + 8)) + grp * 8;
      A.u[t] = *(const unsigned int*)(wRow + kc + ka);
      // B fragment: lanes 0-15 hold K {0..7,16..23}, lanes 16-31 {8..15,24..31}
      int kb = ((t < 4) ? (2 * t) : (2 * t + 8)) + grp * 8;
      B.u[t] = *(const unsigned int*)(bRow + kc + kb);
    }
    acc = __builtin_amdgcn_wmma_f32_16x16x32_f16(false, A.h, false, B.h,
                                                 (short)0, acc, false, false);
  }

  // Epilogue: D layout = VGPR j -> (M=j, lanes 0-15) / (M=j+8, lanes 16-31)
#pragma unroll
  for (int j = 0; j < 8; ++j) {
    int ch = co0 + coTile * 16 + grp * 8 + j;
    int px = x0 + pixTile * 16 + m;
    if (px < W) {
      float s    = gw[ch] * rsqrtf(vw[ch] + EPSBN);
      float bias = bw[ch] - mw[ch] * s;
      float r    = acc[j] * s + bias;
      out[((size_t)(n * Co + ch) * H + h) * W + px] = r > 0.f ? r : 0.f;
    }
  }
}

// ---------------------------------------------------------------------------
// Direct conv3x3 + BN + ReLU for the tiny early layers (Ci<=8, Co=8)
// ---------------------------------------------------------------------------
__global__ void k_conv3_direct(const float* __restrict__ x, const float* __restrict__ w,
                               const float* __restrict__ gw, const float* __restrict__ bw,
                               const float* __restrict__ mw, const float* __restrict__ vw,
                               float* __restrict__ out,
                               int N, int Ci, int H, int W, int Co) {
  size_t i = (size_t)blockIdx.x * blockDim.x + threadIdx.x;
  size_t total = (size_t)N * Co * H * W;
  if (i >= total) return;
  int px = (int)(i % W); size_t t = i / W;
  int py = (int)(t % H); t /= H;
  int co = (int)(t % Co);
  int n  = (int)(t / Co);
  float sum = 0.f;
  for (int ci = 0; ci < Ci; ++ci)
    for (int kh = 0; kh < 3; ++kh) {
      int yy = py + kh - 1;
      if (yy < 0 || yy >= H) continue;
      for (int kw = 0; kw < 3; ++kw) {
        int xx = px + kw - 1;
        if (xx < 0 || xx >= W) continue;
        sum += x[((size_t)(n * Ci + ci) * H + yy) * W + xx] *
               w[((size_t)(co * Ci + ci) * 3 + kh) * 3 + kw];
      }
    }
  float s = gw[co] * rsqrtf(vw[co] + EPSBN);
  float r = sum * s + (bw[co] - mw[co] * s);
  out[i] = r > 0.f ? r : 0.f;
}

// ---------------------------------------------------------------------------
__global__ void k_maxpool2(const float* __restrict__ x, float* __restrict__ out,
                           int N, int C, int H, int W) {
  int Ho = H >> 1, Wo = W >> 1;
  size_t i = (size_t)blockIdx.x * blockDim.x + threadIdx.x;
  size_t total = (size_t)N * C * Ho * Wo;
  if (i >= total) return;
  int ox = (int)(i % Wo); size_t t = i / Wo;
  int oy = (int)(t % Ho); t /= Ho;
  int c  = (int)(t % C);
  int n  = (int)(t / C);
  const float* p = &x[((size_t)(n * C + c) * H + 2 * oy) * W + 2 * ox];
  out[i] = fmaxf(fmaxf(p[0], p[1]), fmaxf(p[W], p[W + 1]));
}

// ---------------------------------------------------------------------------
__global__ void k_conv1x1_bnrelu(const float* __restrict__ x, const float* __restrict__ w,
                                 const float* __restrict__ gw, const float* __restrict__ bw,
                                 const float* __restrict__ mw, const float* __restrict__ vw,
                                 float* __restrict__ out, int N, int Ci, int HW, int Co) {
  size_t i = (size_t)blockIdx.x * blockDim.x + threadIdx.x;
  size_t total = (size_t)N * Co * HW;
  if (i >= total) return;
  int p  = (int)(i % HW); size_t t = i / HW;
  int co = (int)(t % Co);
  int n  = (int)(t / Co);
  float sum = 0.f;
  for (int ci = 0; ci < Ci; ++ci)
    sum += x[((size_t)(n * Ci + ci)) * HW + p] * w[co * Ci + ci];
  float s = gw[co] * rsqrtf(vw[co] + EPSBN);
  float r = sum * s + (bw[co] - mw[co] * s);
  out[i] = r > 0.f ? r : 0.f;
}

__global__ void k_conv1x1_bias(const float* __restrict__ x, const float* __restrict__ w,
                               const float* __restrict__ b, float* __restrict__ out,
                               int N, int Ci, int HW, int Co) {
  size_t i = (size_t)blockIdx.x * blockDim.x + threadIdx.x;
  size_t total = (size_t)N * Co * HW;
  if (i >= total) return;
  int p  = (int)(i % HW); size_t t = i / HW;
  int co = (int)(t % Co);
  int n  = (int)(t / Co);
  float sum = b[co];
  for (int ci = 0; ci < Ci; ++ci)
    sum += x[((size_t)(n * Ci + ci)) * HW + p] * w[co * Ci + ci];
  out[i] = sum;
}

// ---------------------------------------------------------------------------
// SCNN sequential message passing.  One block per batch image.  The current
// and previous lines live in LDS ping-pong buffers so freshly produced data
// is never re-read through global memory.  axis==2: sweep over H, conv along
// W (1x5 kernel).  axis==3: sweep over W, conv along H (5x1 kernel).
// ---------------------------------------------------------------------------
__global__ void k_durl(float* __restrict__ x, const float* __restrict__ w,
                       int C, int H, int W, int axis, int rev) {
  __shared__ float wsm[8 * 8 * 5];
  __shared__ float lA[8 * 100];
  __shared__ float lB[8 * 100];
  int b   = blockIdx.x;
  int tid = threadIdx.x;
  int S = (axis == 2) ? H : W;   // sequence length
  int L = (axis == 2) ? W : H;   // line length
  for (int i = tid; i < C * C * 5; i += blockDim.x) wsm[i] = w[i];

  float* prev = lA;
  float* cur  = lB;
  int i0 = rev ? (S - 1) : 0;
  for (int t = tid; t < C * L; t += blockDim.x) {
    int c = t / L, pos = t - c * L;
    size_t gi = (axis == 2)
        ? (((size_t)(b * C + c) * H + i0) * W + pos)
        : (((size_t)(b * C + c) * H + pos) * W + i0);
    prev[t] = x[gi];
  }
  __syncthreads();

  int stp = rev ? -1 : 1;
  for (int i = i0 + stp; i >= 0 && i < S; i += stp) {
    for (int t = tid; t < C * L; t += blockDim.x) {
      int c = t / L, pos = t - c * L;
      float sum = 0.f;
      for (int ci = 0; ci < C; ++ci) {
        const float* pr = &prev[ci * L];
        const float* wr = &wsm[(c * C + ci) * 5];
#pragma unroll
        for (int u = 0; u < 5; ++u) {
          int pp = pos + u - 2;
          if (pp >= 0 && pp < L) sum += pr[pp] * wr[u];
        }
      }
      size_t gi = (axis == 2)
          ? (((size_t)(b * C + c) * H + i) * W + pos)
          : (((size_t)(b * C + c) * H + pos) * W + i);
      float nv = x[gi] + fmaxf(sum, 0.f);
      cur[t] = nv;
      x[gi]  = nv;
    }
    __syncthreads();
    float* tmp = prev; prev = cur; cur = tmp;
  }
}

// ---------------------------------------------------------------------------
// Bilinear x8 upsample, align-corners mapping: src = o*(n_in-1)/(n_out-1)
// ---------------------------------------------------------------------------
__global__ void k_upsample8(const float* __restrict__ x, float* __restrict__ out,
                            int N, int C, int H, int W) {
  int Ho = H * 8, Wo = W * 8;
  size_t i = (size_t)blockIdx.x * blockDim.x + threadIdx.x;
  size_t total = (size_t)N * C * Ho * Wo;
  if (i >= total) return;
  int ox = (int)(i % Wo); size_t t = i / Wo;
  int oy = (int)(t % Ho); t /= Ho;
  int c  = (int)(t % C);
  int n  = (int)(t / C);
  float sh = oy * (float)(H - 1) / (float)(Ho - 1);
  int   i0 = (int)floorf(sh);
  if (i0 < 0) i0 = 0; if (i0 > H - 2) i0 = H - 2;
  float wh = sh - (float)i0;
  float sw = ox * (float)(W - 1) / (float)(Wo - 1);
  int   j0 = (int)floorf(sw);
  if (j0 < 0) j0 = 0; if (j0 > W - 2) j0 = W - 2;
  float ww = sw - (float)j0;
  const float* p = &x[((size_t)(n * C + c) * H + i0) * W + j0];
  float v0 = p[0] * (1.f - ww) + p[1] * ww;
  float v1 = p[W] * (1.f - ww) + p[W + 1] * ww;
  out[i] = v0 * (1.f - wh) + v1 * wh;
}

// ---------------------------------------------------------------------------
__global__ void k_softmax_avgpool(const float* __restrict__ x, float* __restrict__ out,
                                  int N, int C, int H, int W) {
  int Ho = H >> 1, Wo = W >> 1;
  int i = blockIdx.x * blockDim.x + threadIdx.x;
  if (i >= N * Ho * Wo) return;
  int ow = i % Wo; int t = i / Wo;
  int oy = t % Ho; int n = t / Ho;
  float acc[8];
  for (int c = 0; c < C; ++c) acc[c] = 0.f;
  for (int dy = 0; dy < 2; ++dy)
    for (int dx = 0; dx < 2; ++dx) {
      int yy = 2 * oy + dy, xx = 2 * ow + dx;
      float v[8], mx = -1e30f;
      for (int c = 0; c < C; ++c) {
        v[c] = x[((size_t)(n * C + c) * H + yy) * W + xx];
        mx = fmaxf(mx, v[c]);
      }
      float s = 0.f;
      for (int c = 0; c < C; ++c) { v[c] = expf(v[c] - mx); s += v[c]; }
      float inv = 1.f / s;
      for (int c = 0; c < C; ++c) acc[c] += v[c] * inv;
    }
  for (int c = 0; c < C; ++c)
    out[((size_t)(n * C + c) * Ho + oy) * Wo + ow] = acc[c] * 0.25f;
}

// ---------------------------------------------------------------------------
// FC: one wave32 per output element, lane-strided dot + shfl_xor reduction
// ---------------------------------------------------------------------------
__global__ void k_fc_relu(const float* __restrict__ in, const float* __restrict__ w,
                          const float* __restrict__ bias, float* __restrict__ out,
                          int B, int K, int O) {
  int wv   = (blockIdx.x * blockDim.x + threadIdx.x) >> 5;
  int lane = threadIdx.x & 31;
  if (wv >= B * O) return;
  int b = wv / O, o = wv - b * O;
  float s = 0.f;
  for (int k = lane; k < K; k += 32) s += in[(size_t)b * K + k] * w[(size_t)o * K + k];
  for (int off = 16; off > 0; off >>= 1) s += __shfl_xor(s, off, 32);
  if (lane == 0) out[wv] = fmaxf(s + bias[o], 0.f);
}

__global__ void k_fc_sigmoid(const float* __restrict__ in, const float* __restrict__ w,
                             const float* __restrict__ bias, float* __restrict__ out,
                             int B, int K, int O) {
  int i = blockIdx.x * blockDim.x + threadIdx.x;
  if (i >= B * O) return;
  int b = i / O, o = i - b * O;
  float s = bias[o];
  for (int k = 0; k < K; ++k) s += in[(size_t)b * K + k] * w[(size_t)o * K + k];
  out[i] = 1.f / (1.f + expf(-s));
}

// ---------------------------------------------------------------------------
extern "C" void kernel_launch(void* const* d_in, const int* in_sizes, int n_in,
                              void* d_out, int out_size, void* d_ws, size_t ws_size,
                              hipStream_t stream) {
  (void)in_sizes; (void)n_in; (void)out_size; (void)ws_size;
  const int N = 16;
  const float* x_in = (const float*)d_in[0];

  char* ws = (char*)d_ws;
  float*    bufA = (float*)(ws);                       // 117,964,800 B
  float*    bufB = (float*)(ws + 117964800ull);        // 117,964,800 B
  _Float16* wpk  = (_Float16*)(ws + 235929600ull);     //     262,144 B
  float*    xo   = (float*)(ws + 236191744ull);        //   1,152,000 B
  float*    pbuf = (float*)(ws + 237343744ull);        //     288,000 B
  float*    fcb  = (float*)(ws + 237631744ull);        //       8,192 B

  float* segOut = (float*)d_out;                 // 16*5*288*800
  float* exiOut = segOut + 18432000;             // 16*4

  auto cdiv = [](long long a, long long b) { return (unsigned)((a + b - 1) / b); };

  // WMMA conv layer: repack weights then implicit-GEMM conv
  auto conv_wmma = [&](const float* in, int wi, float* outp,
                       int Ci, int Co, int H, int W, int dil) {
    int Kpad = ((Ci * 9 + 31) / 32) * 32;
    const float* w = (const float*)d_in[wi];
    k_pack_w<<<cdiv((long long)Co * Kpad, 256), 256, 0, stream>>>(w, wpk, Co, Ci, Kpad);
    int nWb = (W + 31) / 32;
    int coBlks = (Co + 63) / 64;
    int bs = ((Co < 64 ? Co : 64) / 16) * 64;   // (coTiles x 2 pixTiles) waves
    dim3 grid((unsigned)(nWb * coBlks), (unsigned)H, (unsigned)N);
    k_conv_wmma<<<grid, bs, 0, stream>>>(
        in, wpk,
        (const float*)d_in[wi + 1], (const float*)d_in[wi + 2],
        (const float*)d_in[wi + 3], (const float*)d_in[wi + 4],
        outp, Ci, H, W, Co, dil, Kpad, nWb);
  };

  auto conv_direct = [&](const float* in, int wi, float* outp,
                         int Ci, int Co, int H, int W) {
    k_conv3_direct<<<cdiv((long long)N * Co * H * W, 256), 256, 0, stream>>>(
        in, (const float*)d_in[wi],
        (const float*)d_in[wi + 1], (const float*)d_in[wi + 2],
        (const float*)d_in[wi + 3], (const float*)d_in[wi + 4],
        outp, N, Ci, H, W, Co);
  };

  auto pool = [&](const float* in, float* outp, int C, int H, int W) {
    k_maxpool2<<<cdiv((long long)N * C * (H / 2) * (W / 2), 256), 256, 0, stream>>>(
        in, outp, N, C, H, W);
  };

  // ---- encoder (enc_params[i] at d_in[1+5i .. 5+5i]) ----
  conv_direct(x_in, 1,  bufA, 3, 8, 288, 800);            // L0
  conv_direct(bufA, 6,  bufB, 8, 8, 288, 800);            // L1
  pool(bufB, bufA, 8, 288, 800);                          // -> 144x400
  conv_wmma(bufA, 11, bufB,  8, 16, 144, 400, 1);         // L2
  conv_wmma(bufB, 16, bufA, 16, 16, 144, 400, 1);         // L3
  pool(bufA, bufB, 16, 144, 400);                         // -> 72x200
  conv_wmma(bufB, 21, bufA, 16, 32, 72, 200, 1);          // L4
  conv_wmma(bufA, 26, bufB, 32, 32, 72, 200, 1);          // L5
  conv_wmma(bufB, 31, bufA, 32, 32, 72, 200, 1);          // L6
  pool(bufA, bufB, 32, 72, 200);                          // -> 36x100
  conv_wmma(bufB, 36, bufA, 32, 64, 36, 100, 1);          // L7
  conv_wmma(bufA, 41, bufB, 64, 64, 36, 100, 1);          // L8
  conv_wmma(bufB, 46, bufA, 64, 64, 36, 100, 1);          // L9
  conv_wmma(bufA, 51, bufB, 64, 64, 36, 100, 1);          // L10
  conv_wmma(bufB, 56, bufA, 64, 64, 36, 100, 1);          // L11
  conv_wmma(bufA, 61, bufB, 64, 64, 36, 100, 1);          // L12

  // ---- pre: dilated 3x3 (64->128, dil=4) then 1x1 (128->8) ----
  conv_wmma(bufB, 66, bufA, 64, 128, 36, 100, 4);         // pre conv1 (w,g,b,m,v at 66..70)
  k_conv1x1_bnrelu<<<cdiv((long long)N * 8 * 3600, 256), 256, 0, stream>>>(
      bufA, (const float*)d_in[71],
      (const float*)d_in[72], (const float*)d_in[73],
      (const float*)d_in[74], (const float*)d_in[75],
      bufB, N, 128, 3600, 8);                             // pre conv2

  // ---- SCNN message passing, in-place on bufB (16,8,36,100) ----
  k_durl<<<N, 256, 0, stream>>>(bufB, (const float*)d_in[76], 8, 36, 100, 2, 0);
  k_durl<<<N, 256, 0, stream>>>(bufB, (const float*)d_in[77], 8, 36, 100, 2, 1);
  k_durl<<<N, 256, 0, stream>>>(bufB, (const float*)d_in[78], 8, 36, 100, 3, 0);
  k_durl<<<N, 256, 0, stream>>>(bufB, (const float*)d_in[79], 8, 36, 100, 3, 1);

  // ---- head: 1x1 (8->5) + bias ----
  k_conv1x1_bias<<<cdiv((long long)N * 5 * 3600, 256), 256, 0, stream>>>(
      bufB, (const float*)d_in[80], (const float*)d_in[81], xo, N, 8, 3600, 5);

  // ---- seg branch: bilinear x8 -> d_out[0 .. 18431999] ----
  k_upsample8<<<cdiv((long long)N * 5 * 288 * 800, 256), 256, 0, stream>>>(
      xo, segOut, N, 5, 36, 100);

  // ---- exi branch: softmax -> avgpool2 -> fc1(relu) -> fc2(sigmoid) ----
  k_softmax_avgpool<<<cdiv((long long)N * 18 * 50, 256), 256, 0, stream>>>(
      xo, pbuf, N, 5, 36, 100);
  // 16*128 outputs, one wave each -> 2048 waves = 65536 threads
  k_fc_relu<<<256, 256, 0, stream>>>(pbuf, (const float*)d_in[82],
                                     (const float*)d_in[83], fcb, N, 4500, 128);
  k_fc_sigmoid<<<1, 64, 0, stream>>>(fcb, (const float*)d_in[84],
                                     (const float*)d_in[85], exiOut, N, 128, 4);
}
